// MultiHeadAttention_86706799771728
// MI455X (gfx1250) — compile-verified, matrix operand order checked by reading, not run on hardware
//
#include <hip/hip_runtime.h>
#include <hip/hip_bf16.h>
#include <math.h>

// MI455X (gfx1250) wave32: MHA with QK-norm.
// Matrix ops: v_wmma_f32_16x16x32_f16 (fp32 accumulate).
// Data movement: global_load_async_to_lds_b128 (ASYNCcnt) double-buffered.

typedef __attribute__((ext_vector_type(16))) _Float16 v16h;
typedef __attribute__((ext_vector_type(4)))  _Float16 v4h;
typedef __attribute__((ext_vector_type(8)))  float    v8f;

#define BATCH  8
#define SEQ    1024
#define DMODEL 1024
#define NHEAD  16
#define HDIM   64
#define MTOT   (BATCH * SEQ)

#define BM 128
#define BN 64
#define KT 32
#define APAD 8   // LDS row padding (halves)

__device__ __forceinline__ v8f wmma_f16(v16h a, v16h b, v8f c) {
  return __builtin_amdgcn_wmma_f32_16x16x32_f16(false, a, false, b, (short)0, c,
                                                false, false);
}

// Async DMA: 16 bytes global -> LDS per lane (tracked by ASYNCcnt).
__device__ __forceinline__ void async_copy16(unsigned lds_addr, const _Float16* gptr) {
  unsigned long long ga = (unsigned long long)(uintptr_t)gptr;
  asm volatile("global_load_async_to_lds_b128 %0, %1, off"
               :: "v"(lds_addr), "v"(ga) : "memory");
}
__device__ __forceinline__ void wait_async() {
  asm volatile("s_wait_asynccnt 0" ::: "memory");
}
// Low 32 bits of a flat pointer to LDS == LDS byte offset (shared aperture).
__device__ __forceinline__ unsigned lds_addr_of(const void* p) {
  return (unsigned)(uintptr_t)p;
}

// 16x32 f16 fragment from LDS, CDNA5 16-bit A-matrix layout:
//   lanes 0-15 : row = row0+lane,    K = {k0..k0+7, k0+16..k0+23}
//   lanes 16-31: row = row0+lane-16, K = {k0+8..k0+15, k0+24..k0+31}
__device__ __forceinline__ v16h load_frag(const _Float16* base, int ld, int row0,
                                          int k0, int lane) {
  const int l = lane & 15;
  const int hsel = (lane >> 4) & 1;
  const _Float16* p = base + (size_t)(row0 + l) * ld + k0 + hsel * 8;
  v16h r;
#pragma unroll
  for (int i = 0; i < 8; ++i) { r[i] = p[i]; r[i + 8] = p[i + 16]; }
  return r;
}

// ---- Precision-conversion passes (hoisted out of the hot loops) ----

__global__ __launch_bounds__(256)
void cvt_f16_kernel(const float* __restrict__ src, _Float16* __restrict__ dst) {
  int i = blockIdx.x * 256 + threadIdx.x;          // one float4 per thread
  float4 v = ((const float4*)src)[i];
  v4h h = { (_Float16)v.x, (_Float16)v.y, (_Float16)v.z, (_Float16)v.w };
  *(v4h*)(dst + (size_t)i * 4) = h;
}

// W fp32 [k][n] -> Wt f16 [n][k] (tiled transpose, both sides coalesced)
__global__ __launch_bounds__(256)
void transpose_cvt_kernel(const float* __restrict__ W, _Float16* __restrict__ Wt) {
  __shared__ _Float16 t[64][65];
  const int tid = threadIdx.x;
  const int n0 = blockIdx.x * 64, k0 = blockIdx.y * 64;
  for (int i = tid; i < 64 * 64; i += 256) {
    int r = i >> 6, c = i & 63;                    // r: k, c: n
    t[c][r] = (_Float16)W[(size_t)(k0 + r) * DMODEL + n0 + c];
  }
  __syncthreads();
  for (int i = tid; i < 64 * 64; i += 256) {
    int r = i >> 6, c = i & 63;                    // r: n, c: k
    Wt[(size_t)(n0 + r) * DMODEL + k0 + c] = t[r][c];
  }
}

// ---- f16 GEMM with fused epilogues ----
// A: f16 [M, DMODEL] row-major.  Bt: f16 [N, K] (= W^T) row-major.
// MODE 0: fp32 out + bias ([M, D], output projection).
// MODE 1: bias + per-head L2 row norm, f16 -> [B,H,S,hd]   (Q, K).
// MODE 2: bias, f16 -> [B,H,hd,S]  (V pre-transposed for P@V).
template <int MODE>
__global__ __launch_bounds__(256)
void gemm_f16_kernel(const _Float16* __restrict__ A, const _Float16* __restrict__ Bt,
                     const float* __restrict__ bias,
                     float* __restrict__ outF, _Float16* __restrict__ outH) {
  __shared__ _Float16 As[2][BM][KT + APAD];
  __shared__ _Float16 Bs[2][BN][KT + APAD];
  __shared__ float    Cs[BM][BN + 1];
  __shared__ float    invn[BM];

  const int tid = threadIdx.x;
  const int wave = tid >> 5, lane = tid & 31;
  const int m0 = blockIdx.x * BM;
  const int n0 = blockIdx.y * BN;
  const int wm = (wave >> 1) * 32;   // 4x2 wave grid of 32x32 regions
  const int wn = (wave & 1) * 32;

  const unsigned asBase = lds_addr_of(&As[0][0][0]);
  const unsigned bsBase = lds_addr_of(&Bs[0][0][0]);
  const unsigned asBuf = (unsigned)sizeof(As[0]);
  const unsigned bsBuf = (unsigned)sizeof(Bs[0]);

  auto stage = [&](int k0, int buf) {
    // A tile: BM x 32 halves = BM*4 16B chunks
    for (int c = tid; c < BM * 4; c += 256) {
      int r = c >> 2, o = (c & 3) * 8;
      async_copy16(asBase + (unsigned)buf * asBuf + (unsigned)((r * (KT + APAD) + o) * 2),
                   A + (size_t)(m0 + r) * DMODEL + k0 + o);
    }
    // B tile: BN x 32 halves
    for (int c = tid; c < BN * 4; c += 256) {
      int r = c >> 2, o = (c & 3) * 8;
      async_copy16(bsBase + (unsigned)buf * bsBuf + (unsigned)((r * (KT + APAD) + o) * 2),
                   Bt + (size_t)(n0 + r) * DMODEL + k0 + o);
    }
  };

  v8f acc[2][2] = {};
  stage(0, 0);
  wait_async();
  __syncthreads();

  const int NK = DMODEL / KT;
  for (int k = 0; k < NK; ++k) {
    const int buf = k & 1;
    if (k + 1 < NK) stage((k + 1) * KT, buf ^ 1);   // prefetch next tile via DMA

    v16h bf0 = load_frag(&Bs[buf][0][0], KT + APAD, wn, 0, lane);
    v16h bf1 = load_frag(&Bs[buf][0][0], KT + APAD, wn + 16, 0, lane);
#pragma unroll
    for (int mt = 0; mt < 2; ++mt) {
      v16h a = load_frag(&As[buf][0][0], KT + APAD, wm + mt * 16, 0, lane);
      acc[mt][0] = wmma_f16(a, bf0, acc[mt][0]);
      acc[mt][1] = wmma_f16(a, bf1, acc[mt][1]);
    }
    wait_async();
    __syncthreads();
  }

  // C fragment -> LDS. value i of lane L: row = i + 8*(L>>4), col = L&15
#pragma unroll
  for (int mt = 0; mt < 2; ++mt)
#pragma unroll
    for (int nt = 0; nt < 2; ++nt)
#pragma unroll
      for (int i = 0; i < 8; ++i) {
        int r = wm + mt * 16 + i + 8 * (lane >> 4);
        int c = wn + nt * 16 + (lane & 15);
        Cs[r][c] = acc[mt][nt][i];
      }
  __syncthreads();

  if constexpr (MODE == 0) {
    for (int i = tid; i < BM * BN; i += 256) {
      int r = i >> 6, c = i & 63;
      outF[(size_t)(m0 + r) * DMODEL + n0 + c] = Cs[r][c] + bias[n0 + c];
    }
  } else {
    if constexpr (MODE == 1) {
      if (tid < BM) {
        float s = 0.f;
        for (int c = 0; c < BN; ++c) {
          float v = Cs[tid][c] + bias[n0 + c];
          s += v * v;
        }
        invn[tid] = 1.f / (sqrtf(s) + 1e-6f);
      }
      __syncthreads();
    }
    const int h = blockIdx.y;  // BN == HDIM: one head per N tile
    for (int i = tid; i < BM * BN; i += 256) {
      int r = i >> 6, c = i & 63;
      int gm = m0 + r;
      int b = gm >> 10, s = gm & (SEQ - 1);
      float v = Cs[r][c] + bias[n0 + c];
      if constexpr (MODE == 1) {
        v *= invn[r];
        outH[(((size_t)b * NHEAD + h) * SEQ + s) * HDIM + c] = (_Float16)v;
      } else {  // V stored transposed: [B,H,hd,S]
        outH[(((size_t)b * NHEAD + h) * HDIM + c) * SEQ + s] = (_Float16)v;
      }
    }
  }
}

// ---- Flash attention ----
// qh/kh: f16 [B,H,S,hd]; vt: f16 [B,H,hd,S]; ctx out: f16 [B,S,D].
// One block per (b,h, 64-query tile); K/V tiles double-buffered via async DMA.
__global__ __launch_bounds__(256)
void flash_attn_kernel(const _Float16* __restrict__ qh, const _Float16* __restrict__ kh,
                       const _Float16* __restrict__ vt, const float* __restrict__ qk_scale,
                       _Float16* __restrict__ ctx) {
  __shared__ _Float16 Qs[64][HDIM + APAD];
  __shared__ _Float16 Ksb[2][64][HDIM + APAD];   // [key][hd]
  __shared__ _Float16 Vtb[2][64][64 + APAD];     // [hd][key]
  __shared__ _Float16 Ps[64][64 + APAD];
  __shared__ float    Ss[64][65];
  __shared__ float    mrow[64], lrow[64], arow[64];

  const int tid = threadIdx.x;
  const int wave = tid >> 5, lane = tid & 31;
  const int q0 = blockIdx.x * 64;
  const int bh = blockIdx.y;
  const size_t headOff = (size_t)bh * SEQ * HDIM;  // same element count for vt
  const float scale = qk_scale[0];
  const int LDK = HDIM + APAD;

  const unsigned qsBase = lds_addr_of(&Qs[0][0]);
  const unsigned ksBase = lds_addr_of(&Ksb[0][0][0]);
  const unsigned vtBase = lds_addr_of(&Vtb[0][0][0]);
  const unsigned ksBuf = (unsigned)sizeof(Ksb[0]);
  const unsigned vtBuf = (unsigned)sizeof(Vtb[0]);

  auto stage_kv = [&](int j0, int buf) {
    for (int c = tid; c < 64 * 8; c += 256) {      // K tile: 64 rows x 64 halves
      int r = c >> 3, o = (c & 7) * 8;
      async_copy16(ksBase + (unsigned)buf * ksBuf + (unsigned)((r * LDK + o) * 2),
                   kh + headOff + (size_t)(j0 + r) * HDIM + o);
    }
    for (int c = tid; c < 64 * 8; c += 256) {      // V^T tile: 64 hd-rows x 64 keys
      int r = c >> 3, o = (c & 7) * 8;
      async_copy16(vtBase + (unsigned)buf * vtBuf + (unsigned)((r * LDK + o) * 2),
                   vt + headOff + (size_t)r * SEQ + j0 + o);
    }
  };

  // Stage Q (once) + first K/V tile
  for (int c = tid; c < 64 * 8; c += 256) {
    int r = c >> 3, o = (c & 7) * 8;
    async_copy16(qsBase + (unsigned)((r * LDK + o) * 2),
                 qh + headOff + (size_t)(q0 + r) * HDIM + o);
  }
  stage_kv(0, 0);
  if (tid < 64) { mrow[tid] = -3.0e38f; lrow[tid] = 0.f; }
  wait_async();
  __syncthreads();

  const int tr = wave >> 1;  // output rows [16*tr, +16)
  const int wc = wave & 1;   // output cols [32*wc, +32)
  v8f o[2] = {};

  for (int j = 0; j < SEQ / 64; ++j) {
    const int buf = j & 1;
    if (j + 1 < SEQ / 64) stage_kv((j + 1) * 64, buf ^ 1);

    // S = Q @ K^T (contract over hd = 64)
    v8f s_acc[2] = {};
#pragma unroll
    for (int kk = 0; kk < HDIM; kk += 32) {
      v16h a  = load_frag(&Qs[0][0], LDK, 16 * tr, kk, lane);
      v16h b0 = load_frag(&Ksb[buf][0][0], LDK, 32 * wc, kk, lane);
      v16h b1 = load_frag(&Ksb[buf][0][0], LDK, 32 * wc + 16, kk, lane);
      s_acc[0] = wmma_f16(a, b0, s_acc[0]);
      s_acc[1] = wmma_f16(a, b1, s_acc[1]);
    }
#pragma unroll
    for (int t = 0; t < 2; ++t)
#pragma unroll
      for (int i = 0; i < 8; ++i) {
        int r = 16 * tr + i + 8 * (lane >> 4);
        int c = 32 * wc + 16 * t + (lane & 15);
        Ss[r][c] = s_acc[t][i] * scale;
      }
    __syncthreads();

    // Online softmax: one thread per query row
    if (tid < 64) {
      float m_old = mrow[tid];
      float mx = m_old;
      for (int c = 0; c < 64; ++c) mx = fmaxf(mx, Ss[tid][c]);
      float alpha = __expf(m_old - mx);
      float sum = 0.f;
      for (int c = 0; c < 64; ++c) {
        float p = __expf(Ss[tid][c] - mx);
        Ps[tid][c] = (_Float16)p;
        sum += p;
      }
      mrow[tid] = mx;
      lrow[tid] = lrow[tid] * alpha + sum;
      arow[tid] = alpha;
    }
    __syncthreads();

    // Rescale accumulator, then O += P @ V (contract over 64 keys)
#pragma unroll
    for (int t = 0; t < 2; ++t)
#pragma unroll
      for (int i = 0; i < 8; ++i)
        o[t][i] *= arow[16 * tr + i + 8 * (lane >> 4)];

#pragma unroll
    for (int kk = 0; kk < 64; kk += 32) {
      v16h a  = load_frag(&Ps[0][0], LDK, 16 * tr, kk, lane);
      v16h b0 = load_frag(&Vtb[buf][0][0], LDK, 32 * wc, kk, lane);
      v16h b1 = load_frag(&Vtb[buf][0][0], LDK, 32 * wc + 16, kk, lane);
      o[0] = wmma_f16(a, b0, o[0]);
      o[1] = wmma_f16(a, b1, o[1]);
    }
    wait_async();
    __syncthreads();
  }

  const int h = bh & (NHEAD - 1), b = bh >> 4;
#pragma unroll
  for (int t = 0; t < 2; ++t)
#pragma unroll
    for (int i = 0; i < 8; ++i) {
      int r = 16 * tr + i + 8 * (lane >> 4);
      int c = 32 * wc + 16 * t + (lane & 15);
      ctx[((size_t)b * SEQ + q0 + r) * DMODEL + h * HDIM + c] =
          (_Float16)(o[t][i] / lrow[r]);
    }
}

extern "C" void kernel_launch(void* const* d_in, const int* in_sizes, int n_in,
                              void* d_out, int out_size, void* d_ws, size_t ws_size,
                              hipStream_t stream) {
  (void)in_sizes; (void)n_in; (void)out_size; (void)ws_size;
  const float* x  = (const float*)d_in[0];
  const float* Wq = (const float*)d_in[1];
  const float* bq = (const float*)d_in[2];
  const float* Wk = (const float*)d_in[3];
  const float* bk = (const float*)d_in[4];
  const float* Wv = (const float*)d_in[5];
  const float* bv = (const float*)d_in[6];
  const float* Wo = (const float*)d_in[7];
  const float* bo = (const float*)d_in[8];
  const float* qk = (const float*)d_in[9];
  float* out = (float*)d_out;

  // Workspace map (f16): xh 16M | Wt_{q,k,v,o} 4x2M | qh 16M | kh 16M | vt 16M | ctx 16M = 88 MiB
  char* ws = (char*)d_ws;
  const size_t MB = 1024 * 1024;
  _Float16* xh   = (_Float16*)(ws);
  _Float16* Wtq  = (_Float16*)(ws + 16 * MB);
  _Float16* Wtk  = (_Float16*)(ws + 18 * MB);
  _Float16* Wtv  = (_Float16*)(ws + 20 * MB);
  _Float16* Wto  = (_Float16*)(ws + 22 * MB);
  _Float16* qh   = (_Float16*)(ws + 24 * MB);
  _Float16* kh   = (_Float16*)(ws + 40 * MB);
  _Float16* vt   = (_Float16*)(ws + 56 * MB);
  _Float16* ctxh = (_Float16*)(ws + 72 * MB);

  cvt_f16_kernel<<<(MTOT * DMODEL / 4) / 256, 256, 0, stream>>>(x, xh);
  dim3 tg(DMODEL / 64, DMODEL / 64);
  transpose_cvt_kernel<<<tg, 256, 0, stream>>>(Wq, Wtq);
  transpose_cvt_kernel<<<tg, 256, 0, stream>>>(Wk, Wtk);
  transpose_cvt_kernel<<<tg, 256, 0, stream>>>(Wv, Wtv);
  transpose_cvt_kernel<<<tg, 256, 0, stream>>>(Wo, Wto);

  dim3 gg(MTOT / BM, DMODEL / BN);
  gemm_f16_kernel<1><<<gg, 256, 0, stream>>>(xh, Wtq, bq, nullptr, qh);
  gemm_f16_kernel<1><<<gg, 256, 0, stream>>>(xh, Wtk, bk, nullptr, kh);
  gemm_f16_kernel<2><<<gg, 256, 0, stream>>>(xh, Wtv, bv, nullptr, vt);
  flash_attn_kernel<<<dim3(SEQ / 64, BATCH * NHEAD), 256, 0, stream>>>(qh, kh, vt, qk, ctxh);
  gemm_f16_kernel<0><<<gg, 256, 0, stream>>>(ctxh, Wto, bo, out, nullptr);
}
